// _lstm_forecaster_90623809945629
// MI455X (gfx1250) — matrix operand order repaired and checked by reading, not compiled, not run on hardware
//
#include <hip/hip_runtime.h>

#define HDIM  64
#define GATES 256
#define SEQ   96
#define PRED  24
#define NB    8192
#define WAVES 4
#define THREADS (WAVES * 32)
#define ROWS_PER_WG (WAVES * 16)

typedef __attribute__((ext_vector_type(16))) __bf16 v16bf;
typedef __attribute__((ext_vector_type(8)))  float  v8f;
typedef __attribute__((ext_vector_type(4)))  float  f32x4;

union Frag { v16bf v; f32x4 q[2]; };

// float -> bf16, round-to-nearest-even
__device__ __forceinline__ unsigned short f2bf(float x) {
  unsigned u = __builtin_bit_cast(unsigned, x);
  u += 0x7FFFu + ((u >> 16) & 1u);
  return (unsigned short)(u >> 16);
}

__device__ __forceinline__ float sigm(float x) {
  return 1.0f / (1.0f + __expf(-x));
}

__device__ __forceinline__ float tanh_f(float x) {
#if __has_builtin(__builtin_amdgcn_tanhf)
  return __builtin_amdgcn_tanhf(x);   // v_tanh_f32 (CDNA5 TRANS op)
#else
  float e = __expf(2.0f * x);
  return 1.0f - 2.0f / (e + 1.0f);
#endif
}

// A-fragment (16-bit 16x32, ISA 7.12.2): lane holds row m = lane%16;
// lanes 0-15 hold K = kf*32 + {0..7, 16..23}, lanes 16-31 K = +8.
__device__ __forceinline__ v16bf load_fragA(const unsigned short* h, int m, int kf, int hi) {
  Frag F;
  const unsigned short* p = h + m * HDIM + kf * 32 + hi * 8;
  F.q[0] = *(const f32x4*)(p);        // ds_load_b128: K pairs for v0..v3
  F.q[1] = *(const f32x4*)(p + 16);   // ds_load_b128: K pairs for v4..v7
  return F.v;
}

// B-fragment (16-bit 32x16): lane holds col n = lane%16;
// lanes 0-15 hold K = kf*32 + {0..7, 16..23}, lanes 16-31 K = +8.
__device__ __forceinline__ v16bf load_fragB(const unsigned short* w, int n, int kf, int hi) {
  Frag F;
  const unsigned short* p = w + n * HDIM + kf * 32 + hi * 8;
  F.q[0] = *(const f32x4*)(p);
  F.q[1] = *(const f32x4*)(p + 16);
  return F.v;
}

__device__ __forceinline__ v8f wmma_bf16(v16bf a, v16bf b, v8f c) {
  return __builtin_amdgcn_wmma_f32_16x16x32_bf16(
      /*neg_a=*/false, a, /*neg_b=*/false, b,
      /*c_mod=*/(short)0, c, /*reuse_a=*/false, /*reuse_b=*/false);
}

__global__ __launch_bounds__(THREADS, 1)
void lstm_fused_kernel(const float* __restrict__ ts,     // (8192, 96, 1)
                       const float* __restrict__ Wih0,   // (256, 1)
                       const float* __restrict__ Whh0,   // (256, 64)
                       const float* __restrict__ bih0,
                       const float* __restrict__ bhh0,
                       const float* __restrict__ Wih1,   // (256, 64)
                       const float* __restrict__ Whh1,   // (256, 64)
                       const float* __restrict__ bih1,
                       const float* __restrict__ bhh1,
                       const float* __restrict__ head_w, // (1, 64)
                       const float* __restrict__ head_b, // (1,)
                       float* __restrict__ out)          // (8192, 24, 1)
{
  extern __shared__ char smem[];
  unsigned short* w0  = (unsigned short*)smem;        // Whh0 bf16 [256][64]
  unsigned short* wi1 = w0  + GATES * HDIM;           // Wih1 bf16 [256][64]
  unsigned short* wh1 = wi1 + GATES * HDIM;           // Whh1 bf16 [256][64]
  unsigned short* hbase = wh1 + GATES * HDIM;         // per-wave h0/h1 (bf16, 16x64 each)

  const int tid  = threadIdx.x;
  const int wave = tid >> 5;
  const int lane = tid & 31;
  const int ln   = lane & 15;   // N (col) / M (row for A) index within tile
  const int hi   = lane >> 4;   // lane-half selector

  // ---- one-time weight staging: fp32 global -> bf16 LDS ----
  for (int i = tid; i < GATES * HDIM; i += THREADS) {
    w0[i]  = f2bf(Whh0[i]);
    wi1[i] = f2bf(Wih1[i]);
    wh1[i] = f2bf(Whh1[i]);
  }
  unsigned short* h0l = hbase + wave * (2 * 16 * HDIM);
  unsigned short* h1l = h0l + 16 * HDIM;
  for (int i = lane; i < 16 * HDIM; i += 32) { h0l[i] = 0; h1l[i] = 0; }
  __syncthreads();

  // ---- per-lane constants (column ln of each 16-wide tile) ----
  float w0c[16], b0c[16], b1c[16];
#pragma unroll
  for (int j = 0; j < 16; ++j) {
    int n = j * 16 + ln;
    w0c[j] = Wih0[n];                 // input weight (input dim = 1)
    b0c[j] = bih0[n] + bhh0[n];
    b1c[j] = bih1[n] + bhh1[n];
  }
  float hwc[4];
#pragma unroll
  for (int jj = 0; jj < 4; ++jj) hwc[jj] = head_w[jj * 16 + ln];
  const float hb = head_b[0];

  const int row0 = blockIdx.x * ROWS_PER_WG + wave * 16;

  v8f c0[4] = {};   // layer-0 cell state, D-layout (rows m = hi*8 + r, cols jj*16+ln)
  v8f c1[4] = {};   // layer-1 cell state

  for (int t = 0; t < SEQ; ++t) {
    // x for this lane's 8 rows (rows m = hi*8 + r)
    float xr[8];
#pragma unroll
    for (int r = 0; r < 8; ++r) xr[r] = ts[(row0 + hi * 8 + r) * SEQ + t];

    // h0_{t-1} A-fragments (grab before overwriting h0 LDS)
    v16bf A0p0 = load_fragA(h0l, ln, 0, hi);
    v16bf A0p1 = load_fragA(h0l, ln, 1, hi);

    // ---------------- layer 0 ----------------
#pragma unroll
    for (int jj = 0; jj < 4; ++jj) {   // hidden-column block (cols jj*16..jj*16+15)
      v8f z[4];
#pragma unroll
      for (int g = 0; g < 4; ++g) {    // gate i/f/g/o
        const int j = g * 4 + jj;      // gate-column tile (256 cols = 16 tiles)
        v8f acc;
#pragma unroll
        for (int r = 0; r < 8; ++r) acc[r] = xr[r] * w0c[j] + b0c[j];  // x-path + biases
        acc = wmma_bf16(A0p0, load_fragB(w0, j * 16 + ln, 0, hi), acc);
        acc = wmma_bf16(A0p1, load_fragB(w0, j * 16 + ln, 1, hi), acc);
        z[g] = acc;
      }
#pragma unroll
      for (int r = 0; r < 8; ++r) {
        float iv = sigm(z[0][r]);
        float fv = sigm(z[1][r]);
        float gv = tanh_f(z[2][r]);
        float ov = sigm(z[3][r]);
        float cv = fv * c0[jj][r] + iv * gv;
        c0[jj][r] = cv;
        float hv = ov * tanh_f(cv);
        h0l[(hi * 8 + r) * HDIM + jj * 16 + ln] = f2bf(hv);  // D-layout -> row-major LDS
      }
    }

    // fragments for layer 1 (same-wave LDS ops are in-order: no barrier needed)
    v16bf A1x0 = load_fragA(h0l, ln, 0, hi);   // h0_t   (layer-1 input)
    v16bf A1x1 = load_fragA(h0l, ln, 1, hi);
    v16bf A1h0 = load_fragA(h1l, ln, 0, hi);   // h1_{t-1}
    v16bf A1h1 = load_fragA(h1l, ln, 1, hi);

    // ---------------- layer 1 ----------------
    float pout[8];
#pragma unroll
    for (int r = 0; r < 8; ++r) pout[r] = 0.0f;

#pragma unroll
    for (int jj = 0; jj < 4; ++jj) {
      v8f z[4];
#pragma unroll
      for (int g = 0; g < 4; ++g) {
        const int j = g * 4 + jj;
        v8f acc;
#pragma unroll
        for (int r = 0; r < 8; ++r) acc[r] = b1c[j];
        acc = wmma_bf16(A1x0, load_fragB(wi1, j * 16 + ln, 0, hi), acc);
        acc = wmma_bf16(A1x1, load_fragB(wi1, j * 16 + ln, 1, hi), acc);
        acc = wmma_bf16(A1h0, load_fragB(wh1, j * 16 + ln, 0, hi), acc);
        acc = wmma_bf16(A1h1, load_fragB(wh1, j * 16 + ln, 1, hi), acc);
        z[g] = acc;
      }
#pragma unroll
      for (int r = 0; r < 8; ++r) {
        float iv = sigm(z[0][r]);
        float fv = sigm(z[1][r]);
        float gv = tanh_f(z[2][r]);
        float ov = sigm(z[3][r]);
        float cv = fv * c1[jj][r] + iv * gv;
        c1[jj][r] = cv;
        float hv = ov * tanh_f(cv);
        h1l[(hi * 8 + r) * HDIM + jj * 16 + ln] = f2bf(hv);
        pout[r] += hv * hwc[jj];   // head partial dot (per-lane column ln of block jj)
      }
    }

    // ---------------- head (last 24 steps) ----------------
    if (t >= SEQ - PRED) {
#pragma unroll
      for (int r = 0; r < 8; ++r) {
        float p = pout[r];
        p += __shfl_xor(p, 1);   // reduce across the 16-lane N dimension
        p += __shfl_xor(p, 2);   // (masks < 16 stay within each lane half)
        p += __shfl_xor(p, 4);
        p += __shfl_xor(p, 8);
        if (ln == 0) {
          out[(row0 + hi * 8 + r) * PRED + (t - (SEQ - PRED))] = p + hb;
        }
      }
    }
  }
}

extern "C" void kernel_launch(void* const* d_in, const int* in_sizes, int n_in,
                              void* d_out, int out_size, void* d_ws, size_t ws_size,
                              hipStream_t stream) {
  (void)in_sizes; (void)n_in; (void)out_size; (void)d_ws; (void)ws_size;
  const float* ts     = (const float*)d_in[0];
  const float* Wih0   = (const float*)d_in[1];
  const float* Whh0   = (const float*)d_in[2];
  const float* bih0   = (const float*)d_in[3];
  const float* bhh0   = (const float*)d_in[4];
  const float* Wih1   = (const float*)d_in[5];
  const float* Whh1   = (const float*)d_in[6];
  const float* bih1   = (const float*)d_in[7];
  const float* bhh1   = (const float*)d_in[8];
  const float* head_w = (const float*)d_in[9];
  const float* head_b = (const float*)d_in[10];
  float* out = (float*)d_out;

  // LDS: 3 bf16 weight matrices (96 KB) + 4 waves * (h0+h1) bf16 buffers (16 KB)
  // CDNA5 WGP has 320 KB LDS, so 112 KB/WG is fine (2 WGs/WGP).
  size_t smem = (size_t)3 * GATES * HDIM * sizeof(unsigned short)
              + (size_t)WAVES * 2 * 16 * HDIM * sizeof(unsigned short);

  lstm_fused_kernel<<<NB / ROWS_PER_WG, THREADS, smem, stream>>>(
      ts, Wih0, Whh0, bih0, bhh0, Wih1, Whh1, bih1, bhh1, head_w, head_b, out);
}
